// Audition_36979668418928
// MI455X (gfx1250) — matrix operand order repaired and checked by reading, not compiled, not run on hardware
//
#include <hip/hip_runtime.h>
#include <math.h>

// ---------------- problem constants (from reference) ----------------
#define T_STEPS 1000
#define BATCH   64
#define INP     80
#define KPAD    96      // INP padded to multiple of 32 for bf16 WMMA
#define HID     1024
#define NWG     16      // persistent workgroups; each owns HID/NWG = 64 columns
#define LEAK    0.5f

typedef __attribute__((ext_vector_type(16))) __bf16        v16bf;
typedef __attribute__((ext_vector_type(8)))  float         v8f;
typedef __attribute__((ext_vector_type(8)))  unsigned int  v8u;

static __device__ __forceinline__ unsigned short f2bf(float f) {
  return __builtin_bit_cast(unsigned short, (__bf16)f);
}

// ---- A operand (16x32 bf16, M x K): lane L -> M = L&15, half = L>>4.
// ISA table: vgpr j holds K pair at  half*8 + (j<4 ? 0 : 16) + 2*(j&3).
// Two contiguous 16B chunks per lane -> 2x global_load_b128.
static __device__ __forceinline__ v16bf load_a_row(const unsigned short* __restrict__ row,
                                                   int k0, int half) {
  const unsigned short* rp = row + k0 + half * 8;
  v8u d;
#pragma unroll
  for (int j = 0; j < 8; ++j) {
    int off = ((j < 4) ? 0 : 16) + ((j & 3) * 2);
    d[j] = *(const unsigned int*)(rp + off);
  }
  return __builtin_bit_cast(v16bf, d);
}

// Fallback A operand built on-the-fly from fp32 x row (K padded: k >= INP -> 0)
static __device__ __forceinline__ v16bf load_a_x(const float* __restrict__ row,
                                                 int k0, int half) {
  int kb = k0 + half * 8;
  v8u d;
#pragma unroll
  for (int j = 0; j < 8; ++j) {
    int k = kb + ((j < 4) ? 0 : 16) + ((j & 3) * 2);
    float f0 = (k     < INP) ? row[k]     : 0.0f;
    float f1 = (k + 1 < INP) ? row[k + 1] : 0.0f;
    d[j] = (unsigned int)f2bf(f0) | ((unsigned int)f2bf(f1) << 16);
  }
  return __builtin_bit_cast(v16bf, d);
}

// ---- B operand (32x16 bf16, K x N): lane L -> N = L&15; same K order as A:
// vgpr j holds K pair at half*8 + (j<4 ? 0 : 16) + 2*(j&3).
static __device__ __forceinline__ v16bf load_b_row(const unsigned short* __restrict__ row,
                                                   int k0, int half) {
  const unsigned short* rp = row + k0 + half * 8;
  v8u d;
#pragma unroll
  for (int j = 0; j < 8; ++j) d[j] = *(const unsigned int*)(rp + ((j < 4) ? 0 : 16) + ((j & 3) * 2));
  return __builtin_bit_cast(v16bf, d);
}

// ---------------- init: reset barrier, h0 = 0, convert weights (+x) to bf16 ----------------
__global__ void esn_init(const float* __restrict__ whh, const float* __restrict__ wih,
                         const float* __restrict__ x,
                         unsigned short* __restrict__ whh_bf,
                         unsigned short* __restrict__ wih_bf,
                         unsigned short* __restrict__ x_bf,     // may be null
                         unsigned short* __restrict__ h16,
                         unsigned int* cnt, int do_x) {
  size_t i = (size_t)blockIdx.x * blockDim.x + threadIdx.x;
  size_t stride = (size_t)gridDim.x * blockDim.x;
  if (i == 0) *cnt = 0u;
  for (size_t k = i; k < (size_t)HID * HID; k += stride) whh_bf[k] = f2bf(whh[k]);
  for (size_t k = i; k < (size_t)HID * KPAD; k += stride) {
    size_t r = k / KPAD, c = k % KPAD;
    wih_bf[k] = (c < INP) ? f2bf(wih[r * INP + c]) : (unsigned short)0;
  }
  for (size_t k = i; k < (size_t)BATCH * HID; k += stride) h16[k] = 0;  // h0, buffer 0
  if (do_x) {
    for (size_t k = i; k < (size_t)T_STEPS * BATCH * KPAD; k += stride) {
      size_t r = k / KPAD, c = k % KPAD;
      x_bf[k] = (c < INP) ? f2bf(x[r * INP + c]) : (unsigned short)0;
    }
  }
}

// ---------------- persistent scan: 16 WGs x 256 threads (8 waves) ----------------
template <bool XPRE>
__global__ __launch_bounds__(256) void esn_scan(
    const float* __restrict__ x, const unsigned short* __restrict__ x_bf,
    const int* __restrict__ lengths,
    float* __restrict__ out,
    const unsigned short* __restrict__ whh_bf,
    const unsigned short* __restrict__ wih_bf,
    unsigned short* __restrict__ h16,
    unsigned int* cnt) {
  const int g    = blockIdx.x;          // column-slice owner: cols [g*64, g*64+64)
  const int tid  = threadIdx.x;
  const int w    = tid >> 5;            // wave 0..7
  const int lane = tid & 31;
  const int r    = w & 3;               // batch-tile row (16 rows each)
  const int q    = w >> 2;              // column-pair (2 x 16 cols each)
  const int half = lane >> 4;
  const int ln   = lane & 15;

  const int mRow = r * 16 + ln;                 // A-operand batch row for this lane
  const int col0 = g * 64 + q * 32 + ln;        // tile-0 output column
  const int col1 = col0 + 16;                   // tile-1 output column

  // C-fragment batch rows (fixed across time): vgpr e -> M = r*16 + half*8 + e
  int brow[8], len8[8];
#pragma unroll
  for (int e = 0; e < 8; ++e) {
    brow[e] = r * 16 + half * 8 + e;
    len8[e] = lengths[brow[e]];
  }

  // Leaky fp32 state lives in registers: each lane reads back exactly the
  // (b, col) positions it wrote last step, so no global fp32 h is needed.
  float hreg0[8], hreg1[8];
#pragma unroll
  for (int e = 0; e < 8; ++e) { hreg0[e] = 0.0f; hreg1[e] = 0.0f; }

  const unsigned short* wihR0 = wih_bf + (size_t)col0 * KPAD;
  const unsigned short* wihR1 = wih_bf + (size_t)col1 * KPAD;
  const unsigned short* whhR0 = whh_bf + (size_t)col0 * HID;
  const unsigned short* whhR1 = whh_bf + (size_t)col1 * HID;

  int p = 0;  // bf16 h double-buffer parity: read p, write p^1
  for (int t = 0; t < T_STEPS; ++t) {
    v8f acc0 = {}; v8f acc1 = {};

    // -------- input drive: x[t] @ W_ih^T  (K = 96, zero-padded) --------
    if (XPRE) {
      const unsigned short* xrow = x_bf + ((size_t)t * BATCH + mRow) * KPAD;
      if (t + 1 < T_STEPS)
        __builtin_prefetch(xrow + BATCH * KPAD, 0, 1);  // global_prefetch_b8
#pragma unroll
      for (int kc = 0; kc < KPAD / 32; ++kc) {
        v16bf a  = load_a_row(xrow, kc * 32, half);
        v16bf b0 = load_b_row(wihR0, kc * 32, half);
        v16bf b1 = load_b_row(wihR1, kc * 32, half);
        acc0 = __builtin_amdgcn_wmma_f32_16x16x32_bf16(false, a, false, b0, (short)0, acc0, false, false);
        acc1 = __builtin_amdgcn_wmma_f32_16x16x32_bf16(false, a, false, b1, (short)0, acc1, false, false);
      }
    } else {
      const float* xrow = x + ((size_t)t * BATCH + mRow) * INP;
      if (t + 1 < T_STEPS)
        __builtin_prefetch(xrow + BATCH * INP, 0, 1);    // global_prefetch_b8
#pragma unroll
      for (int kc = 0; kc < KPAD / 32; ++kc) {
        v16bf a  = load_a_x(xrow, kc * 32, half);
        v16bf b0 = load_b_row(wihR0, kc * 32, half);
        v16bf b1 = load_b_row(wihR1, kc * 32, half);
        acc0 = __builtin_amdgcn_wmma_f32_16x16x32_bf16(false, a, false, b0, (short)0, acc0, false, false);
        acc1 = __builtin_amdgcn_wmma_f32_16x16x32_bf16(false, a, false, b1, (short)0, acc1, false, false);
      }
    }

    // -------- recurrent drive: h @ W_hh^T  (K = 1024, L2-resident bf16) --------
    const unsigned short* hrow = h16 + (size_t)p * (BATCH * HID) + (size_t)mRow * HID;
#pragma unroll 4
    for (int kc = 0; kc < HID / 32; ++kc) {
      v16bf a  = load_a_row(hrow, kc * 32, half);
      v16bf b0 = load_b_row(whhR0, kc * 32, half);
      v16bf b1 = load_b_row(whhR1, kc * 32, half);
      acc0 = __builtin_amdgcn_wmma_f32_16x16x32_bf16(false, a, false, b0, (short)0, acc0, false, false);
      acc1 = __builtin_amdgcn_wmma_f32_16x16x32_bf16(false, a, false, b1, (short)0, acc1, false, false);
    }

    // -------- epilogue: leak blend + tanh + length mask, write ys[t] & h' --------
    unsigned short* hn16 = h16 + (size_t)(p ^ 1) * (BATCH * HID);
    float*          outt = out + (size_t)t * (BATCH * HID);
#pragma unroll
    for (int e = 0; e < 8; ++e) {
      const int b = brow[e];
      const bool m = t < len8[e];
      {
        float nh  = (1.0f - LEAK) * hreg0[e] + LEAK * tanhf(acc0[e]);
        float res = m ? nh : hreg0[e];
        hreg0[e]  = res;
        hn16[(size_t)b * HID + col0] = f2bf(res);
        outt[(size_t)b * HID + col0] = m ? nh : 0.0f;
      }
      {
        float nh  = (1.0f - LEAK) * hreg1[e] + LEAK * tanhf(acc1[e]);
        float res = m ? nh : hreg1[e];
        hreg1[e]  = res;
        hn16[(size_t)b * HID + col1] = f2bf(res);
        outt[(size_t)b * HID + col1] = m ? nh : 0.0f;
      }
    }

    // -------- device-wide barrier (persistent grid, monotone counter) --------
    __threadfence();
    __syncthreads();
    if (tid == 0) {
      atomicAdd(cnt, 1u);
      const unsigned target = (unsigned)(NWG * (t + 1));
      volatile unsigned int* vc = cnt;
      while (*vc < target) __builtin_amdgcn_s_sleep(1);
    }
    __syncthreads();
    __threadfence();
    p ^= 1;
  }
}

// ---------------- host launcher ----------------
extern "C" void kernel_launch(void* const* d_in, const int* in_sizes, int n_in,
                              void* d_out, int out_size, void* d_ws, size_t ws_size,
                              hipStream_t stream) {
  (void)in_sizes; (void)n_in; (void)out_size;
  const float* x       = (const float*)d_in[0];   // [T,B,I]
  const int*   lengths = (const int*)d_in[1];     // [B]
  const float* W_ih    = (const float*)d_in[2];   // [H,I]
  const float* W_hh    = (const float*)d_in[3];   // [H,H]
  float*       out     = (float*)d_out;           // [T,B,H]

  // workspace layout (all offsets 256B aligned)
  const size_t SZ_H16  = (size_t)2 * BATCH * HID * sizeof(unsigned short);   // 256 KB
  const size_t SZ_WHH  = (size_t)HID * HID * sizeof(unsigned short);         // 2 MB
  const size_t SZ_WIH  = (size_t)HID * KPAD * sizeof(unsigned short);        // 192 KB
  const size_t SZ_XBF  = (size_t)T_STEPS * BATCH * KPAD * sizeof(unsigned short); // 12.3 MB

  char* ws = (char*)d_ws;
  unsigned int*   cnt    = (unsigned int*)ws;
  unsigned short* h16    = (unsigned short*)(ws + 256);
  unsigned short* whh_bf = (unsigned short*)((char*)h16 + SZ_H16);
  unsigned short* wih_bf = (unsigned short*)((char*)whh_bf + SZ_WHH);
  unsigned short* x_bf   = (unsigned short*)((char*)wih_bf + SZ_WIH);

  const size_t need_small = 256 + SZ_H16 + SZ_WHH + SZ_WIH;
  const bool   xpre       = ws_size >= need_small + SZ_XBF;  // fixed per run -> deterministic

  esn_init<<<512, 256, 0, stream>>>(W_hh, W_ih, x, whh_bf, wih_bf,
                                    xpre ? x_bf : (unsigned short*)nullptr,
                                    h16, cnt, xpre ? 1 : 0);
  if (xpre)
    esn_scan<true ><<<NWG, 256, 0, stream>>>(x, x_bf, lengths, out, whh_bf, wih_bf, h16, cnt);
  else
    esn_scan<false><<<NWG, 256, 0, stream>>>(x, x_bf, lengths, out, whh_bf, wih_bf, h16, cnt);
}
